// RealVectorSHT_43989055045856
// MI455X (gfx1250) — matrix operand order repaired and checked by reading, not compiled, not run on hardware
//
#include <hip/hip_runtime.h>

// ---------------------------------------------------------------------------
// Real vector SHT on MI455X (gfx1250), bf16-WMMA, f32 accumulate.
//   init A : build bf16 twiddle table (736x736, zero padded) in d_ws
//   init B : zero the K-pad lanes of the bf16 intermediate
//   Stage 1: rFFT over longitude as GEMM vs. twiddle table
//            (46208 rows x K=720 x N=722) -> bf16 xf[comp][reim][m][bc][kpad=384]
//   Stage 2: per-m contraction: 8 GEMMs (M=64, N=361 l, K=361 lat) fused as
//            4 A-frags x 2 B-frags -> 8 v_wmma per K-step, signs folded at store.
// Roofline: ~0.78 GB @ 23.3 TB/s ~= 33us ; ~100 GFLOP bf16 WMMA ~= 20us.
// d_ws requirement: 70,975,488 (xf) + 1,083,392 (twiddles) = 72,058,880 bytes.
// ---------------------------------------------------------------------------

typedef __attribute__((ext_vector_type(16))) __bf16 v16bf;
typedef __attribute__((ext_vector_type(8)))  __bf16 v8bf;
typedef __attribute__((ext_vector_type(8)))  float  v8f;
typedef float f32x4u __attribute__((ext_vector_type(4), aligned(4))); // dword-aligned b128

#define NLATC 361
#define NLONC 720
#define MMAXC 361
#define LMAXC 361
#define NCOL1 722                  // [re m=0..360 | im m=0..360]
#define KPAD2 384                  // stage-2 K (=nlat) padded to 12*32
#define WSTR2 24576                // 64 * KPAD2, elems per (comp,reim,m) slice
#define PLANE2 8871936             // 361 * WSTR2
#define WSLICE 130321              // LMAXC * NLATC (f32 weights, per m)
#define WPLANE 47045881            // MMAXC * LMAXC * NLATC
#define TWID_COLS 736              // 46*16 (N padded)
#define TWID_K 736                 // 23*32 (K padded)
#define XF_ELEMS 35487744          // 4 * PLANE2

__device__ __forceinline__ v8f wmma_bf16(v16bf a, v16bf b, v8f c) {
    return __builtin_amdgcn_wmma_f32_16x16x32_bf16(false, a, false, b,
                                                   (short)0, c, false, false);
}

// A-fragment layout: two 8-element K-runs at +0 and +16 (ISA 16-bit A 16x32).
__device__ __forceinline__ v16bf ld_a_bf16(const __bf16* p) {
    union { v16bf v; v8bf h[2]; } u;
    u.h[0] = *reinterpret_cast<const v8bf*>(p);
    u.h[1] = *reinterpret_cast<const v8bf*>(p + 16);
    return u.v;
}
// B-fragment layout: one contiguous 16-element K-run (k = 16h + s order, not the ISA table order).
__device__ __forceinline__ v16bf ld_b_bf16(const __bf16* p) {
    union { v16bf v; v8bf h[2]; } u;
    u.h[0] = *reinterpret_cast<const v8bf*>(p);
    u.h[1] = *reinterpret_cast<const v8bf*>(p + 8);
    return u.v;
}
// Convert 16 consecutive f32 (dword-aligned) -> B fragment.
__device__ __forceinline__ v16bf ld_b_f32cvt(const float* p) {
    f32x4u q0 = *reinterpret_cast<const f32x4u*>(p);
    f32x4u q1 = *reinterpret_cast<const f32x4u*>(p + 4);
    f32x4u q2 = *reinterpret_cast<const f32x4u*>(p + 8);
    f32x4u q3 = *reinterpret_cast<const f32x4u*>(p + 12);
    v16bf r;
#pragma unroll
    for (int j = 0; j < 4; ++j) {
        r[j]      = (__bf16)q0[j];
        r[j + 4]  = (__bf16)q1[j];
        r[j + 8]  = (__bf16)q2[j];
        r[j + 12] = (__bf16)q3[j];
    }
    return r;
}

// ============ init A: bf16 twiddle table [col][k], zero padded =============
__global__ void __launch_bounds__(256)
sht_twiddle_init(__bf16* __restrict__ twid) {
    const float scale = 6.28318530717958647692f / 720.0f; // 2*pi/nlon
    int idx = blockIdx.x * 256 + threadIdx.x;             // grid covers 736*736
    int col = idx / TWID_K;
    int k   = idx % TWID_K;
    float v = 0.0f;
    if (col < NCOL1 && k < NLONC) {
        int isIm = (col >= MMAXC) ? 1 : 0;
        int m    = isIm ? col - MMAXC : col;
        int ph   = (m * k) % NLONC;                       // exact phase reduction
        float th = (float)ph * scale;
        float sn, cs;
        __sincosf(th, &sn, &cs);
        v = (isIm ? -sn : cs) * scale;
    }
    twid[idx] = (__bf16)v;
}

// ============ init B: zero K-pad region of xf intermediate =================
__global__ void __launch_bounds__(256)
sht_pad_init(__bf16* __restrict__ xf) {
    int row = blockIdx.x * 256 + threadIdx.x;             // 4*361*64 = 92416 rows
    size_t base = (size_t)row * KPAD2;
#pragma unroll
    for (int k = NLATC; k < KPAD2; ++k) xf[base + k] = (__bf16)0.0f;
}

// ======================= Stage 1: DFT (rFFT) as GEMM =======================
__global__ void __launch_bounds__(256)
sht_dft_kernel(const float* __restrict__ x, const __bf16* __restrict__ twid,
               __bf16* __restrict__ xf) {
    const int lane    = threadIdx.x & 31;
    const int wave    = threadIdx.x >> 5;     // 0..7
    const int mt      = wave & 3;             // M-tile within 64-row block
    const int halfSel = lane >> 4;
    const int lrow    = lane & 15;
    const int rowBase = blockIdx.x * 64 + mt * 16;

    // nt-invariant scatter bases for the C-tile rows this thread owns
    size_t rbase[8];
#pragma unroll
    for (int r = 0; r < 8; ++r) {
        int R    = rowBase + r + halfSel * 8;
        int bc   = R / 722;
        int comp = (R % 722) / NLATC;
        int lat  = R % NLATC;
        rbase[r] = (size_t)comp * (2u * PLANE2) + (size_t)bc * KPAD2 + (size_t)lat;
    }
    const float* rp = x + (size_t)(rowBase + lrow) * NLONC; // 16B aligned rows

    for (int nt = (wave >> 2); nt < 46; nt += 2) {
        const int col   = nt * 16 + lrow;                 // 0..735
        const int isIm  = (col >= MMAXC) ? 1 : 0;
        const int mcol  = isIm ? col - MMAXC : col;
        const bool colOK = (col < NCOL1);
        const size_t colpart = (size_t)isIm * PLANE2 + (size_t)mcol * WSTR2;
        const __bf16* tb = twid + (size_t)col * TWID_K;   // padded cols are valid

        v8f acc = {};
        for (int kt = 0; kt < 23; ++kt) {
            const int kb = kt * 32 + halfSel * 8;
            // A fragment: f32 -> bf16, two aligned float4 pairs per K-run.
            v16bf a;
            {
                float4 q0 = *reinterpret_cast<const float4*>(rp + kb);
                float4 q1 = *reinterpret_cast<const float4*>(rp + kb + 4);
                a[0] = (__bf16)q0.x; a[1] = (__bf16)q0.y; a[2] = (__bf16)q0.z; a[3] = (__bf16)q0.w;
                a[4] = (__bf16)q1.x; a[5] = (__bf16)q1.y; a[6] = (__bf16)q1.z; a[7] = (__bf16)q1.w;
                if (kt < 22) {  // uniform: last K-tile's 2nd run is k>=720 for ALL lanes
                    float4 q2 = *reinterpret_cast<const float4*>(rp + kb + 16);
                    float4 q3 = *reinterpret_cast<const float4*>(rp + kb + 20);
                    a[8]  = (__bf16)q2.x; a[9]  = (__bf16)q2.y; a[10] = (__bf16)q2.z; a[11] = (__bf16)q2.w;
                    a[12] = (__bf16)q3.x; a[13] = (__bf16)q3.y; a[14] = (__bf16)q3.z; a[15] = (__bf16)q3.w;
                } else {
#pragma unroll
                    for (int j = 8; j < 16; ++j) a[j] = (__bf16)0.0f;
                }
            }
            // B fragment: precomputed bf16 twiddles, two 8-element K-runs at +0 and +16.
            v16bf bt = ld_a_bf16(tb + kt * 32 + halfSel * 8);
            acc = wmma_bf16(a, bt, acc);
        }
        if (colOK) {
#pragma unroll
            for (int r = 0; r < 8; ++r) xf[rbase[r] + colpart] = (__bf16)acc[r];
        }
    }
}

// ================ Stage 2: per-m contraction (8 fused GEMMs) ===============
__global__ void __launch_bounds__(256)
sht_contract_kernel(const __bf16* __restrict__ xf, const float* __restrict__ w,
                    float* __restrict__ out) {
    const int m       = blockIdx.x;           // 0..360
    const int lane    = threadIdx.x & 31;
    const int wave    = threadIdx.x >> 5;
    const int mt      = wave & 3;             // batch-row tile (M=64 -> 4 tiles)
    const int halfSel = lane >> 4;
    const int lrow    = lane & 15;

    const __bf16* a00 = xf + (size_t)(0 * MMAXC + m) * WSTR2; // comp0 re
    const __bf16* a01 = xf + (size_t)(1 * MMAXC + m) * WSTR2; // comp0 im
    const __bf16* a10 = xf + (size_t)(2 * MMAXC + m) * WSTR2; // comp1 re
    const __bf16* a11 = xf + (size_t)(3 * MMAXC + m) * WSTR2; // comp1 im
    const float*  w0  = w + (size_t)m * WSLICE;
    const float*  w1  = w + (size_t)WPLANE + (size_t)m * WSLICE;

    const int rowoff = (mt * 16 + lrow) * KPAD2;          // A row base (n index)

    for (int nt = blockIdx.y * 2 + (wave >> 2); nt < 23; nt += 4) {
        const int l    = nt * 16 + lrow;
        const int lc   = l < LMAXC ? l : LMAXC - 1;       // clamp; junk cols masked at store
        const bool lok = (l < LMAXC);
        const float* w0p = w0 + (size_t)lc * NLATC;
        const float* w1p = w1 + (size_t)lc * NLATC;

        v8f p1 = {}, p2 = {}, p3 = {}, p4 = {}, p5 = {}, p6 = {}, p7 = {}, p8 = {};
        for (int kt = 0; kt < 12; ++kt) {
            const int kbA = kt * 32 + halfSel * 16;
            // A fragments: unconditional 16B loads (K padded+zeroed to 384)
            v16bf A00 = ld_b_bf16(a00 + rowoff + kbA);
            v16bf A01 = ld_b_bf16(a01 + rowoff + kbA);
            v16bf A10 = ld_b_bf16(a10 + rowoff + kbA);
            v16bf A11 = ld_b_bf16(a11 + rowoff + kbA);
            // B fragments: weights f32 -> bf16 in-register
            const int kbB = kt * 32 + halfSel * 16;
            v16bf Bw0, Bw1;
            if (kt < 11) {
                __builtin_prefetch(w0p + kbB + 32, 0, 1); // next K-tile (1.4KB-strided rows)
                __builtin_prefetch(w1p + kbB + 32, 0, 1);
                Bw0 = ld_b_f32cvt(w0p + kbB);
                Bw1 = ld_b_f32cvt(w1p + kbB);
            } else {            // K tail: 361..383 invalid
#pragma unroll
                for (int j = 0; j < 16; ++j) {
                    int k  = kbB + j;
                    int kc = k < NLATC ? k : NLATC - 1;
                    bool ok = (k < NLATC);
                    float f0 = w0p[kc], f1 = w1p[kc];
                    Bw0[j] = (__bf16)(ok ? f0 : 0.0f);
                    Bw1[j] = (__bf16)(ok ? f1 : 0.0f);
                }
            }
            // 8 WMMAs: {a0re,a0im,a1re,a1im} x {w0,w1}
            p1 = wmma_bf16(A00, Bw0, p1);   // a0re*w0
            p2 = wmma_bf16(A11, Bw1, p2);   // a1im*w1
            p3 = wmma_bf16(A01, Bw0, p3);   // a0im*w0
            p4 = wmma_bf16(A10, Bw1, p4);   // a1re*w1
            p5 = wmma_bf16(A01, Bw1, p5);   // a0im*w1
            p6 = wmma_bf16(A10, Bw0, p6);   // a1re*w0
            p7 = wmma_bf16(A00, Bw1, p7);   // a0re*w1
            p8 = wmma_bf16(A11, Bw0, p8);   // a1im*w0
        }
        // combine signs + store complex pairs
        if (lok) {
#pragma unroll
            for (int r = 0; r < 8; ++r) {
                int n = mt * 16 + r + halfSel * 8;        // batch index 0..63
                float2 s, t;
                s.x =  p1[r] - p2[r];                     // s_re
                s.y =  p3[r] + p4[r];                     // s_im
                t.x = -p5[r] - p6[r];                     // t_re
                t.y =  p7[r] - p8[r];                     // t_im
                size_t bs = ((size_t)(n * 2 + 0) * LMAXC + l) * (size_t)(2 * MMAXC) + 2 * m;
                size_t bt = ((size_t)(n * 2 + 1) * LMAXC + l) * (size_t)(2 * MMAXC) + 2 * m;
                *reinterpret_cast<float2*>(out + bs) = s;
                *reinterpret_cast<float2*>(out + bt) = t;
            }
        }
    }
}

// ============================== launcher ===================================
extern "C" void kernel_launch(void* const* d_in, const int* in_sizes, int n_in,
                              void* d_out, int out_size, void* d_ws, size_t ws_size,
                              hipStream_t stream) {
    const float* x = (const float*)d_in[0];   // (4,16,2,361,720) f32
    const float* w = (const float*)d_in[1];   // (2,361,361,361)  f32
    float* out = (float*)d_out;               // (4,16,2,361,361) complex64 -> f32 pairs

    __bf16* xf   = (__bf16*)d_ws;                                   // 70,975,488 B
    __bf16* twid = (__bf16*)((char*)d_ws + (size_t)XF_ELEMS * 2);   //  1,083,392 B

    sht_twiddle_init<<<dim3((TWID_COLS * TWID_K) / 256), dim3(256), 0, stream>>>(twid);
    sht_pad_init<<<dim3(361), dim3(256), 0, stream>>>(xf);          // 92416 rows
    sht_dft_kernel<<<dim3(722), dim3(256), 0, stream>>>(x, twid, xf);
    sht_contract_kernel<<<dim3(361, 2), dim3(256), 0, stream>>>(xf, w, out);
}
